// BasicQuantumDQN_64579128262688
// MI455X (gfx1250) — matrix operand order repaired and checked by reading, not compiled, not run on hardware
//
#include <hip/hip_runtime.h>

#define N_QUBITS 12
#define DIM      4096          // 2^12 amplitudes
#define N_LAYERS 4
#define N_ACTIONS 6
#define TPB      256           // 8 wave32 per workgroup

typedef __attribute__((ext_vector_type(16))) _Float16 v16h;
typedef __attribute__((ext_vector_type(8)))  float    v8f;

// ---------------------------------------------------------------------------
// LDS bank swizzle: GF(2)-linear remap of the low 5 index bits so that all
// three pass layouts (lane-varying bit sets {4:0}, {8,3:0}, {8:4}) map the 32
// lanes of a wave onto 32 distinct bank pairs -> conflict-free b64 DS ops.
//   p0 = i0^i8, p1 = i1^i5, p2 = i2^i6, p3 = i3^i7, p4 = i4^i8
// ---------------------------------------------------------------------------
__device__ __forceinline__ int swz(int i) {
  int low = (i & 31) ^ ((i >> 4) & 0x0E) ^ (((i >> 8) & 1) ? 0x11 : 0);
  return (i & ~31) | low;
}

// Pass layouts: which 4 index bits live in registers (v = 0..15)
__device__ __forceinline__ int idxA(int tid, int v) {  // local bits 11:8 (wires 0-3)
  return (v << 8) | tid;
}
__device__ __forceinline__ int idxB(int tid, int v) {  // local bits 7:4  (wires 4-7)
  return ((tid & 0xF0) << 4) | (v << 4) | (tid & 0x0F);
}
__device__ __forceinline__ int idxC(int tid, int v) {  // local bits 3:0  (wires 8-11)
  return (tid << 4) | v;
}

// Complex 2x2 butterfly: (a,b) <- U * (a,b)
__device__ __forceinline__ void bfly(float2& a, float2& b,
    float u00r, float u00i, float u01r, float u01i,
    float u10r, float u10i, float u11r, float u11i) {
  float2 na, nb;
  na.x = u00r * a.x - u00i * a.y + u01r * b.x - u01i * b.y;
  na.y = u00r * a.y + u00i * a.x + u01r * b.y + u01i * b.x;
  nb.x = u10r * a.x - u10i * a.y + u11r * b.x - u11i * b.y;
  nb.y = u10r * a.y + u10i * a.x + u11r * b.y + u11i * b.x;
  a = na; b = nb;
}

// Apply 4 consecutive-wire gates to the 16 register-resident amplitudes.
// Wire (base+w) corresponds to local bit (3-w) of v.
__device__ __forceinline__ void apply4(float2 a[16], const float* g4) {
#pragma unroll
  for (int w = 0; w < 4; ++w) {
    const float4 gA = *(const float4*)(g4 + w * 8);      // u00r u00i u01r u01i
    const float4 gB = *(const float4*)(g4 + w * 8 + 4);  // u10r u10i u11r u11i
    const int m = 1 << (3 - w);
#pragma unroll
    for (int v = 0; v < 16; ++v) {
      if (!(v & m))
        bfly(a[v], a[v | m], gA.x, gA.y, gA.z, gA.w, gB.x, gB.y, gB.z, gB.w);
    }
  }
}

__global__ __launch_bounds__(TPB)
void qsim_kernel(const float* __restrict__ x,      // (B, 12)
                 const float* __restrict__ wgt,    // (4, 12, 3)
                 const float* __restrict__ fc_w,   // (6, 12)
                 const float* __restrict__ fc_b,   // (6,)
                 float* __restrict__ out) {        // (B, 6)
  __shared__ float2 st[DIM];                       // statevector, 32 KB
  __shared__ float4 gm4[N_LAYERS * N_QUBITS * 2];  // Rot matrices, 1.5 KB
  __shared__ float  enc_c[N_QUBITS], enc_s[N_QUBITS];
  __shared__ float  qacc[16];

  float* gm = (float*)gm4;
  const int tid = threadIdx.x;
  const int b   = blockIdx.x;

  // ---- setup: Rot matrices, encoding angles, accumulators ----
  if (tid < N_LAYERS * N_QUBITS) {
    const float* w = wgt + tid * 3;
    float phi = w[0], th = w[1], om = w[2];
    float ct = __cosf(0.5f * th), s_t = __sinf(0.5f * th);
    float ap = 0.5f * (phi + om), am = 0.5f * (phi - om);
    float cap = __cosf(ap), sap = __sinf(ap);
    float cam = __cosf(am), sam = __sinf(am);
    float* g = gm + tid * 8;
    g[0] =  cap * ct;  g[1] = -sap * ct;   // u00 = e^{-i(phi+om)/2} cos(th/2)
    g[2] = -cam * s_t; g[3] = -sam * s_t;  // u01 = -e^{ i(phi-om)/2} sin(th/2)
    g[4] =  cam * s_t; g[5] = -sam * s_t;  // u10 =  e^{-i(phi-om)/2} sin(th/2)
    g[6] =  cap * ct;  g[7] =  sap * ct;   // u11 = e^{ i(phi+om)/2} cos(th/2)
  }
  if (tid < N_QUBITS) {
    float h = 0.5f * x[b * N_QUBITS + tid];
    enc_c[tid] = __cosf(h);
    enc_s[tid] = __sinf(h);
  }
  if (tid < 16) qacc[tid] = 0.f;
  __syncthreads();

  float2 a[16];

  // ---- layer 0, pass A: RY-encoded state is a tensor product, build it
  //      directly in registers (no LDS read), then gates on wires 0-3 ----
  {
    float pf = 1.f;                       // product over wires 4..11 (tid bits)
#pragma unroll
    for (int q = 4; q < N_QUBITS; ++q)
      pf *= ((tid >> (11 - q)) & 1) ? enc_s[q] : enc_c[q];
#pragma unroll
    for (int v = 0; v < 16; ++v) {
      float lf = pf;                      // product over wires 0..3 (v bits)
#pragma unroll
      for (int w = 0; w < 4; ++w)
        lf *= ((v >> (3 - w)) & 1) ? enc_s[w] : enc_c[w];
      a[v] = make_float2(lf, 0.f);
    }
    apply4(a, gm);                        // layer 0, wires 0-3
#pragma unroll
    for (int v = 0; v < 16; ++v) st[swz(idxA(tid, v))] = a[v];
  }
  __syncthreads();

  // ---- layers: 3 register-blocked passes each (gates commute in-layer) ----
  for (int l = 0; l < N_LAYERS; ++l) {
    const float* gl = gm + l * (N_QUBITS * 8);

    if (l > 0) {                          // pass A: wires 0-3
#pragma unroll
      for (int v = 0; v < 16; ++v) a[v] = st[swz(idxA(tid, v))];
      apply4(a, gl);
#pragma unroll
      for (int v = 0; v < 16; ++v) st[swz(idxA(tid, v))] = a[v];
      __syncthreads();
    }

    // pass B: wires 4-7
#pragma unroll
    for (int v = 0; v < 16; ++v) a[v] = st[swz(idxB(tid, v))];
    apply4(a, gl + 32);
#pragma unroll
    for (int v = 0; v < 16; ++v) st[swz(idxB(tid, v))] = a[v];
    __syncthreads();

    // pass C: wires 8-11, then fold in the CZ-chain diagonal
#pragma unroll
    for (int v = 0; v < 16; ++v) a[v] = st[swz(idxC(tid, v))];
    apply4(a, gl + 64);
    if (l < N_LAYERS - 1) {
#pragma unroll
      for (int v = 0; v < 16; ++v) {      // CZ sign = parity of adjacent 1-pairs
        int i = idxC(tid, v);
        if (__popc(i & (i >> 1)) & 1) { a[v].x = -a[v].x; a[v].y = -a[v].y; }
      }
#pragma unroll
      for (int v = 0; v < 16; ++v) st[swz(idxC(tid, v))] = a[v];
    } else {
      // final layer: last CZ cannot change probabilities -> skip it and
      // store probabilities in-place (own slots only, race-free)
#pragma unroll
      for (int v = 0; v < 16; ++v)
        st[swz(idxC(tid, v))].x = a[v].x * a[v].x + a[v].y * a[v].y;
    }
    __syncthreads();
  }

  // ---- WMMA readout: q_out[q] = sum_i probs[i] * (1 - 2*bit(i, q)) ----
  // Index split: i = wv*512 + M*32 + k  (k = i[4:0], M = i[8:5], wv = i[11:9]).
  // A (16x32 f16) = probs slice of this wave; B columns 7..11 carry the +-1
  // signs for wires whose bit lives inside k; column 12 is all-ones so
  // D[M][12] = per-chunk probability sums used for the M/wv-level wires.
  {
    const int lane = tid & 31;
    const int wv   = tid >> 5;       // wave32 on gfx1250
    const int M    = lane & 15;
    const int hi   = lane >> 4;
    const int base = wv * 512;

    v16h av;
#pragma unroll
    for (int e = 0; e < 16; ++e) {
      // 16-bit A-matrix layout: lanes 0-15 hold K 0-7 / 16-23, hi lanes +8
      int k = (e & 7) + ((e >> 3) << 4) + (hi << 3);
      av[e] = (_Float16)st[swz(base + M * 32 + k)].x;
    }
    v16h bm;
#pragma unroll
    for (int e = 0; e < 16; ++e) {
      // 16-bit B-matrix layout: lanes 0-15 hold K 0-7 / 16-23, hi lanes +8
      int k = (e & 7) + ((e >> 3) << 4) + (hi << 3);
      float v = 0.f;
      if (M >= 7 && M <= 11) v = ((k >> (11 - M)) & 1) ? -1.f : 1.f; // wire M
      else if (M == 12)      v = 1.f;                                 // row sums
      bm[e] = (_Float16)v;
    }
    v8f acc = {};
    acc = __builtin_amdgcn_wmma_f32_16x16x32_f16(
        /*neg_a=*/false, av, /*neg_b=*/false, bm,
        /*c_mod=*/(short)0, acc, /*reuse_a=*/false, /*reuse_b=*/false);

    // C/D layout: lane n holds column n; VGPR v = row (v + 8*hi).
    float colsum = acc[0] + acc[1] + acc[2] + acc[3] +
                   acc[4] + acc[5] + acc[6] + acc[7];
    if (M >= 7 && M <= 11) atomicAdd(&qacc[M], colsum);   // k-level wires 7..11
    if (M == 12) {                                        // chunk sums
      float tot = 0.f, q3 = 0.f, q4 = 0.f, q5 = 0.f, q6 = 0.f;
#pragma unroll
      for (int v = 0; v < 8; ++v) {
        int   Mc = v + hi * 8;     // chunk index = i[8:5]
        float cs = acc[v];
        tot += cs;
        q3 += ((Mc >> 3) & 1) ? -cs : cs;  // wire 3 -> bit 8 of i
        q4 += ((Mc >> 2) & 1) ? -cs : cs;  // wire 4 -> bit 7
        q5 += ((Mc >> 1) & 1) ? -cs : cs;  // wire 5 -> bit 6
        q6 += ((Mc >> 0) & 1) ? -cs : cs;  // wire 6 -> bit 5
      }
      atomicAdd(&qacc[3], q3);
      atomicAdd(&qacc[4], q4);
      atomicAdd(&qacc[5], q5);
      atomicAdd(&qacc[6], q6);
      // wave-level wires 0..2 -> bits 11..9 of i = bits 2..0 of wv
      atomicAdd(&qacc[0], ((wv >> 2) & 1) ? -tot : tot);
      atomicAdd(&qacc[1], ((wv >> 1) & 1) ? -tot : tot);
      atomicAdd(&qacc[2], ((wv >> 0) & 1) ? -tot : tot);
    }
  }
  __syncthreads();

  // ---- tiny FC: out[b, a] = q_out . fc_w[a] + fc_b[a] ----
  if (tid < N_ACTIONS) {
    float o = fc_b[tid];
#pragma unroll
    for (int q = 0; q < N_QUBITS; ++q) o += qacc[q] * fc_w[tid * N_QUBITS + q];
    out[b * N_ACTIONS + tid] = o;
  }
}

extern "C" void kernel_launch(void* const* d_in, const int* in_sizes, int n_in,
                              void* d_out, int out_size, void* d_ws, size_t ws_size,
                              hipStream_t stream) {
  const float* x    = (const float*)d_in[0];  // (B, 12)
  const float* wgt  = (const float*)d_in[1];  // (4, 12, 3)
  const float* fc_w = (const float*)d_in[2];  // (6, 12)
  const float* fc_b = (const float*)d_in[3];  // (6,)
  float*       out  = (float*)d_out;          // (B, 6)

  const int batch = in_sizes[0] / N_QUBITS;   // 4096
  hipLaunchKernelGGL(qsim_kernel, dim3(batch), dim3(TPB), 0, stream,
                     x, wgt, fc_w, fc_b, out);
}
